// TeamRatingGNN_15676630630999
// MI455X (gfx1250) — matrix operand order repaired and checked by reading, not compiled, not run on hardware
//
#include <hip/hip_runtime.h>
#include <stdint.h>

// ---------------- problem constants (from reference) ----------------
#define N_TEAMS 50000
#define EMBD    64
#define HIDD    128
#define NEDGES  400000
#define NBATCH  16384
#define NPASSES 3

#define XSTR 136   // LDS row stride in halves (128 + 8 pad to stagger banks)

// ---------------- CDNA5 WMMA types ----------------
typedef __attribute__((ext_vector_type(16))) __bf16       v16bf;
typedef __attribute__((ext_vector_type(8)))  float        v8f;
typedef __attribute__((ext_vector_type(4)))  unsigned int u32x4;

struct Frag32B { u32x4 lo, hi; };

__device__ __forceinline__ unsigned short f2bf(float f) {
  unsigned int u = __builtin_bit_cast(unsigned int, f);
  u += 0x7FFFu + ((u >> 16) & 1u);          // round-to-nearest-even
  return (unsigned short)(u >> 16);
}

// A-matrix 16x32 bf16 fragment (contiguous k order, same as B):
// lane: m = lane%16, g = lane/16; VGPR0..7 hold k = 16g+0..15 -> 16 contiguous halves.
__device__ __forceinline__ v16bf load_a_frag(const unsigned short* base) {
  Frag32B f;
  f.lo = *(const u32x4*)(base);
  f.hi = *(const u32x4*)(base + 8);
  return __builtin_bit_cast(v16bf, f);
}

// B-matrix 32x16 bf16 fragment from W^T stored [n][k] (row stride HIDD halves):
// lane: n = lane%16, g = lane/16; VGPR0..7 hold k = 16g+0..15 -> 16 contiguous halves.
__device__ __forceinline__ v16bf load_b_frag(const unsigned short* base) {
  Frag32B f;
  f.lo = *(const u32x4*)(base);
  f.hi = *(const u32x4*)(base + 8);
  return __builtin_bit_cast(v16bf, f);
}

__device__ __forceinline__ v8f wmma_bf16(v16bf a, v16bf b, v8f c) {
  // (neg_a, A, neg_b, B, c_mod, C, reuse_a, reuse_b)
  return __builtin_amdgcn_wmma_f32_16x16x32_bf16(false, a, false, b, (short)0, c,
                                                 false, false);
}

// Async gather: copy 128 bytes (one bf16 row-half) global -> LDS via the CDNA5
// async path (GLOBAL_LOAD_ASYNC_TO_LDS_B128, GV mode, tracked on ASYNCcnt).
__device__ __forceinline__ void stage_row_async(const unsigned short* gsrc,
                                                unsigned lds_byte) {
#pragma unroll
  for (int j = 0; j < 8; ++j) {
    unsigned long long ga = (unsigned long long)(const void*)gsrc + 16ull * j;
    unsigned la = lds_byte + 16u * j;
    asm volatile("global_load_async_to_lds_b128 %0, %1, off"
                 :: "v"(la), "v"(ga) : "memory");
  }
}

__device__ __forceinline__ void wait_async_and_barrier() {
  asm volatile("s_wait_asynccnt 0x0" ::: "memory");
  __syncthreads();
}

// f32 row -> bf16 LDS row-half with scale (used for the aggregate input)
__device__ __forceinline__ void stage_row_scaled(const float* __restrict__ src,
                                                 unsigned short* __restrict__ dst,
                                                 float scale) {
  const float4* s4 = (const float4*)src;
#pragma unroll
  for (int j = 0; j < 16; ++j) {
    float4 v = s4[j];
    unsigned int lo = (unsigned)f2bf(v.x * scale) | ((unsigned)f2bf(v.y * scale) << 16);
    unsigned int hi = (unsigned)f2bf(v.z * scale) | ((unsigned)f2bf(v.w * scale) << 16);
    *(uint2*)(dst + 4 * j) = make_uint2(lo, hi);
  }
}

// ---------------- weight convert: f32 [K][N] -> bf16 transposed [N][K] ----------------
__global__ void gnn_cvt_wT(const float* __restrict__ src,
                           unsigned short* __restrict__ dst, int K, int N) {
  int idx = blockIdx.x * blockDim.x + threadIdx.x;
  if (idx < K * N) {
    int k = idx / N, n = idx % N;
    dst[n * K + k] = f2bf(src[idx]);
  }
}

// ---------------- flat f32 -> bf16 convert (emb -> te) ----------------
__global__ void gnn_cvt_flat(const float* __restrict__ src,
                             unsigned short* __restrict__ dst, int n) {
  int idx = blockIdx.x * blockDim.x + threadIdx.x;
  if (idx < n) dst[idx] = f2bf(src[idx]);
}

// ---------------- in-degree counts (pass invariant) ----------------
__global__ void gnn_count(const int* __restrict__ edges, float* __restrict__ cnt,
                          int nE) {
  int idx = blockIdx.x * blockDim.x + threadIdx.x;
  if (idx < 2 * nE) {
    int ei = (idx < nE) ? idx : idx - nE;
    int r  = (idx < nE) ? edges[2 * ei + 1] : edges[2 * ei];
    atomicAdd(&cnt[r], 1.0f);
  }
}

// ---------------- edge message kernel: 128 edges / block ----------------
// msg = relu([te_s,te_r] @ W1 + b1) @ W2 + b2; atomic segment-sum into sums[recv]
__global__ __launch_bounds__(256) void gnn_msg(
    const unsigned short* __restrict__ te, const int* __restrict__ edges,
    const unsigned short* __restrict__ w1t, const float* __restrict__ b1,
    const unsigned short* __restrict__ w2t, const float* __restrict__ b2,
    float* __restrict__ sums, int nE) {
  __shared__ unsigned short Xs[128 * XSTR];  // X, then reused for H (wave-private rows)
  __shared__ int rid[128];

  int t = threadIdx.x;
  int row = t >> 1, half = t & 1;
  long e = (long)blockIdx.x * 128 + row;
  long total = 2L * nE;

  int gid = 0, rr = -1;
  if (e < total) {
    int fwd = (e < (long)nE);
    int ei  = fwd ? (int)e : (int)(e - nE);
    int a = edges[2 * ei], b = edges[2 * ei + 1];
    int snd = fwd ? a : b;
    rr = fwd ? b : a;
    gid = half ? rr : snd;
  }
  if (!half) rid[row] = rr;
  if (t == 0) { __builtin_prefetch(w1t, 0, 3); __builtin_prefetch(w2t, 0, 3); }

  unsigned xbase = (unsigned)(unsigned long long)(void*)Xs;
  stage_row_async(te + (long)gid * EMBD, xbase + row * (XSTR * 2) + half * 128);
  wait_async_and_barrier();

  int lane = t & 31, wv = t >> 5;
  int lm = lane & 15, lg = lane >> 4;
  int m0 = wv * 16;

  v16bf af[4];
#pragma unroll
  for (int kc = 0; kc < 4; ++kc)
    af[kc] = load_a_frag(Xs + (m0 + lm) * XSTR + kc * 32 + lg * 16);

  // GEMM1: H = relu(X @ W1 + b1)  (128x128 x 128x128) -> back into Xs
#pragma unroll
  for (int nt = 0; nt < 8; ++nt) {
    v8f acc = {0.f, 0.f, 0.f, 0.f, 0.f, 0.f, 0.f, 0.f};
#pragma unroll
    for (int kc = 0; kc < 4; ++kc)
      acc = wmma_bf16(af[kc], load_b_frag(w1t + (nt * 16 + lm) * HIDD + kc * 32 + lg * 16), acc);
    int n = nt * 16 + lm;
    float bias = b1[n];
#pragma unroll
    for (int r = 0; r < 8; ++r) {
      float v = acc[r] + bias;
      v = v > 0.f ? v : 0.f;
      Xs[(m0 + r + 8 * lg) * XSTR + n] = f2bf(v);
    }
  }
  __syncthreads();

  v16bf hf[4];
#pragma unroll
  for (int kc = 0; kc < 4; ++kc)
    hf[kc] = load_a_frag(Xs + (m0 + lm) * XSTR + kc * 32 + lg * 16);

  // GEMM2: M = H @ W2 + b2 (128x128 x 128x64), scatter-add to sums
#pragma unroll
  for (int nt = 0; nt < 4; ++nt) {
    v8f acc = {0.f, 0.f, 0.f, 0.f, 0.f, 0.f, 0.f, 0.f};
#pragma unroll
    for (int kc = 0; kc < 4; ++kc)
      acc = wmma_bf16(hf[kc], load_b_frag(w2t + (nt * 16 + lm) * HIDD + kc * 32 + lg * 16), acc);
    int n = nt * 16 + lm;
    float bias = b2[n];
#pragma unroll
    for (int r = 0; r < 8; ++r) {
      int rcv = rid[m0 + r + 8 * lg];
      if (rcv >= 0) atomicAdd(&sums[(long)rcv * EMBD + n], acc[r] + bias);
    }
  }
}

// ---------------- node update kernel: 128 teams / block ----------------
// upd = relu([te, sums/max(cnt,1)] @ uW + ub); te' = cnt>0 ? upd : te
__global__ __launch_bounds__(256) void gnn_update(
    const unsigned short* __restrict__ te_in, const float* __restrict__ sums,
    const float* __restrict__ cnt, const unsigned short* __restrict__ uwt,
    const float* __restrict__ ub, unsigned short* __restrict__ te_out, int nT) {
  __shared__ unsigned short Xs[128 * XSTR];

  int t = threadIdx.x;
  int row = t >> 1, half = t & 1;
  int team = blockIdx.x * 128 + row;
  int tc = team < nT ? team : nT - 1;

  unsigned xbase = (unsigned)(unsigned long long)(void*)Xs;
  if (!half) {
    stage_row_async(te_in + (long)tc * EMBD, xbase + row * (XSTR * 2));
  } else {
    float inv = 1.0f / fmaxf(cnt[tc], 1.0f);
    stage_row_scaled(sums + (long)tc * EMBD, Xs + row * XSTR + EMBD, inv);
  }
  wait_async_and_barrier();

  int lane = t & 31, wv = t >> 5;
  int lm = lane & 15, lg = lane >> 4;
  int m0 = wv * 16;

  v16bf af[4];
#pragma unroll
  for (int kc = 0; kc < 4; ++kc)
    af[kc] = load_a_frag(Xs + (m0 + lm) * XSTR + kc * 32 + lg * 16);

#pragma unroll
  for (int nt = 0; nt < 4; ++nt) {
    v8f acc = {0.f, 0.f, 0.f, 0.f, 0.f, 0.f, 0.f, 0.f};
#pragma unroll
    for (int kc = 0; kc < 4; ++kc)
      acc = wmma_bf16(af[kc], load_b_frag(uwt + (nt * 16 + lm) * HIDD + kc * 32 + lg * 16), acc);
    int n = nt * 16 + lm;
    float bias = ub[n];
#pragma unroll
    for (int r = 0; r < 8; ++r) {
      int m = m0 + r + 8 * lg;
      int tm = blockIdx.x * 128 + m;
      if (tm < nT) {
        float c = cnt[tm];
        unsigned short o;
        if (c > 0.f) {
          float v = acc[r] + bias;
          o = f2bf(v > 0.f ? v : 0.f);
        } else {
          o = te_in[(long)tm * EMBD + n];
        }
        te_out[(long)tm * EMBD + n] = o;
      }
    }
  }
}

// ---------------- prediction head: 128 games / block ----------------
__global__ __launch_bounds__(256) void gnn_head(
    const unsigned short* __restrict__ te, const int* __restrict__ home,
    const int* __restrict__ away, const unsigned short* __restrict__ p1t,
    const float* __restrict__ pb1, const unsigned short* __restrict__ p2t,
    const float* __restrict__ pb2, const float* __restrict__ pw3,
    const float* __restrict__ pb3, float* __restrict__ out, int nB) {
  __shared__ unsigned short Xs[128 * XSTR];  // X then H (wave-private rows)
  __shared__ float H2[128 * 32];

  int t = threadIdx.x;
  int row = t >> 1, half = t & 1;
  int i = blockIdx.x * 128 + row;
  int gid = 0;
  if (i < nB) gid = half ? away[i] : home[i];

  unsigned xbase = (unsigned)(unsigned long long)(void*)Xs;
  stage_row_async(te + (long)gid * EMBD, xbase + row * (XSTR * 2) + half * 128);
  wait_async_and_barrier();

  int lane = t & 31, wv = t >> 5;
  int lm = lane & 15, lg = lane >> 4;
  int m0 = wv * 16;

  v16bf af[4];
#pragma unroll
  for (int kc = 0; kc < 4; ++kc)
    af[kc] = load_a_frag(Xs + (m0 + lm) * XSTR + kc * 32 + lg * 16);

  // GEMM1: 128x128 x 128x128, relu, back into Xs
#pragma unroll
  for (int nt = 0; nt < 8; ++nt) {
    v8f acc = {0.f, 0.f, 0.f, 0.f, 0.f, 0.f, 0.f, 0.f};
#pragma unroll
    for (int kc = 0; kc < 4; ++kc)
      acc = wmma_bf16(af[kc], load_b_frag(p1t + (nt * 16 + lm) * HIDD + kc * 32 + lg * 16), acc);
    int n = nt * 16 + lm;
    float bias = pb1[n];
#pragma unroll
    for (int r = 0; r < 8; ++r) {
      float v = acc[r] + bias;
      v = v > 0.f ? v : 0.f;
      Xs[(m0 + r + 8 * lg) * XSTR + n] = f2bf(v);
    }
  }
  __syncthreads();

  v16bf hf[4];
#pragma unroll
  for (int kc = 0; kc < 4; ++kc)
    hf[kc] = load_a_frag(Xs + (m0 + lm) * XSTR + kc * 32 + lg * 16);

  // GEMM2: 128x128 x 128x32, relu -> H2 (f32)
#pragma unroll
  for (int nt = 0; nt < 2; ++nt) {
    v8f acc = {0.f, 0.f, 0.f, 0.f, 0.f, 0.f, 0.f, 0.f};
#pragma unroll
    for (int kc = 0; kc < 4; ++kc)
      acc = wmma_bf16(hf[kc], load_b_frag(p2t + (nt * 16 + lm) * HIDD + kc * 32 + lg * 16), acc);
    int n = nt * 16 + lm;
    float bias = pb2[n];
#pragma unroll
    for (int r = 0; r < 8; ++r) {
      float v = acc[r] + bias;
      H2[(m0 + r + 8 * lg) * 32 + n] = v > 0.f ? v : 0.f;
    }
  }
  __syncthreads();

  if (t < 128) {
    int g = blockIdx.x * 128 + t;
    if (g < nB) {
      float s = pb3[0];
#pragma unroll
      for (int j = 0; j < 32; ++j) s += H2[t * 32 + j] * pw3[j];
      out[g] = 1.0f / (1.0f + __expf(-s));
    }
  }
}

// ---------------- host launcher ----------------
extern "C" void kernel_launch(void* const* d_in, const int* in_sizes, int n_in,
                              void* d_out, int out_size, void* d_ws, size_t ws_size,
                              hipStream_t stream) {
  (void)in_sizes; (void)n_in; (void)out_size; (void)ws_size;

  const int*   home = (const int*)d_in[0];
  const int*   away = (const int*)d_in[1];
  const int*   edges = (const int*)d_in[2];
  const float* emb  = (const float*)d_in[3];
  const float* mW1  = (const float*)d_in[4];
  const float* mb1  = (const float*)d_in[5];
  const float* mW2  = (const float*)d_in[6];
  const float* mb2  = (const float*)d_in[7];
  const float* uW   = (const float*)d_in[8];
  const float* ub   = (const float*)d_in[9];
  const float* pW1  = (const float*)d_in[10];
  const float* pb1  = (const float*)d_in[11];
  const float* pW2  = (const float*)d_in[12];
  const float* pb2  = (const float*)d_in[13];
  const float* pW3  = (const float*)d_in[14];
  const float* pb3  = (const float*)d_in[15];

  char* w = (char*)d_ws;
  size_t off = 0;
  auto carve = [&](size_t bytes) -> void* {
    off = (off + 255) & ~(size_t)255;
    void* p = w + off;
    off += bytes;
    return p;
  };

  unsigned short* teA  = (unsigned short*)carve((size_t)N_TEAMS * EMBD * 2);
  unsigned short* teB  = (unsigned short*)carve((size_t)N_TEAMS * EMBD * 2);
  float* sums = (float*)carve((size_t)N_TEAMS * EMBD * sizeof(float));
  float* cnt  = (float*)carve((size_t)N_TEAMS * sizeof(float));
  unsigned short* w1t = (unsigned short*)carve((size_t)HIDD * HIDD * 2);
  unsigned short* w2t = (unsigned short*)carve((size_t)EMBD * HIDD * 2);
  unsigned short* uwt = (unsigned short*)carve((size_t)EMBD * HIDD * 2);
  unsigned short* p1t = (unsigned short*)carve((size_t)HIDD * HIDD * 2);
  unsigned short* p2t = (unsigned short*)carve((size_t)32 * HIDD * 2);

  hipMemsetAsync(cnt, 0, (size_t)N_TEAMS * sizeof(float), stream);

  gnn_cvt_flat<<<(N_TEAMS * EMBD + 255) / 256, 256, 0, stream>>>(emb, teA,
                                                                 N_TEAMS * EMBD);
  gnn_cvt_wT<<<(HIDD * HIDD + 255) / 256, 256, 0, stream>>>(mW1, w1t, HIDD, HIDD);
  gnn_cvt_wT<<<(HIDD * EMBD + 255) / 256, 256, 0, stream>>>(mW2, w2t, HIDD, EMBD);
  gnn_cvt_wT<<<(HIDD * EMBD + 255) / 256, 256, 0, stream>>>(uW, uwt, HIDD, EMBD);
  gnn_cvt_wT<<<(HIDD * HIDD + 255) / 256, 256, 0, stream>>>(pW1, p1t, HIDD, HIDD);
  gnn_cvt_wT<<<(HIDD * 32 + 255) / 256, 256, 0, stream>>>(pW2, p2t, HIDD, 32);
  gnn_count<<<(2 * NEDGES + 255) / 256, 256, 0, stream>>>(edges, cnt, NEDGES);

  unsigned short* cur = teA;
  unsigned short* nxt = teB;
  for (int p = 0; p < NPASSES; ++p) {
    hipMemsetAsync(sums, 0, (size_t)N_TEAMS * EMBD * sizeof(float), stream);
    gnn_msg<<<(2 * NEDGES + 127) / 128, 256, 0, stream>>>(cur, edges, w1t, mb1, w2t,
                                                          mb2, sums, NEDGES);
    gnn_update<<<(N_TEAMS + 127) / 128, 256, 0, stream>>>(cur, sums, cnt, uwt, ub,
                                                          nxt, N_TEAMS);
    unsigned short* tmp = cur; cur = nxt; nxt = tmp;
  }

  gnn_head<<<(NBATCH + 127) / 128, 256, 0, stream>>>(cur, home, away, p1t, pb1, p2t,
                                                     pb2, pW3, pb3, (float*)d_out,
                                                     NBATCH);
}
